// LSTM_86354612453405
// MI455X (gfx1250) — matrix operand order repaired and checked by reading, not compiled, not run on hardware
//
#include <hip/hip_runtime.h>
#include <hip/hip_bf16.h>

// ---------------------------------------------------------------------------
// LSTM forward on CDNA5 (gfx1250), wave32 + v_wmma_f32_16x16x32_bf16.
// gates_t = [x_t | h_{t-1}] @ [Wx ; Wh] + b   (fused, K=1024, bf16 WMMA, f32 acc)
// ---------------------------------------------------------------------------

typedef __attribute__((ext_vector_type(16))) __bf16 v16bf;
typedef __attribute__((ext_vector_type(8)))  float  v8f;

#define Bsz 64
#define Ssz 1024
#define Dsz 512
#define Hsz 512

// ---------------- fp32 -> bf16 conversion of the input tensor ---------------
__global__ __launch_bounds__(256) void conv_bf16(const float* __restrict__ x,
                                                 __bf16* __restrict__ y) {
    size_t i = ((size_t)blockIdx.x * 256 + threadIdx.x) * 4;
    float4 v = *(const float4*)(x + i);
    union { __bf16 h[4]; uint2 u; } o;
    o.h[0] = (__bf16)v.x; o.h[1] = (__bf16)v.y;
    o.h[2] = (__bf16)v.z; o.h[3] = (__bf16)v.w;
    *(uint2*)(y + i) = o.u;
}

// ------------- pack [Wx;Wh] (K=1024 x N=2048) into B-fragment order ---------
// B layout chosen here (32x16 bf16 B, wave32; contiguous k, same as the A loads): lane L -> column n = L&15,
// half = L>>4, element i (0..15) -> k = half*16 + i   (within a 32-row kstep).
// Linear packed index = ((ntile*32 + ks)*32 + lane)*16 + i  == tid bit-fields.
__global__ __launch_bounds__(256) void pack_w(
    const float* __restrict__ Wxi, const float* __restrict__ Wxf,
    const float* __restrict__ Wxo, const float* __restrict__ Wxc,
    const float* __restrict__ Whi, const float* __restrict__ Whf,
    const float* __restrict__ Who, const float* __restrict__ Whc,
    __bf16* __restrict__ Wpack) {
    unsigned tid = blockIdx.x * 256 + threadIdx.x;   // [0, 2^21)
    unsigned e16  = tid & 15;
    unsigned lane = (tid >> 4) & 31;
    unsigned ks   = (tid >> 9) & 31;
    unsigned ntg  = tid >> 14;                       // 0..127
    unsigned n    = ntg * 16 + (lane & 15);          // 0..2047 (gate-major)
    unsigned hlf  = lane >> 4;
    unsigned k    = ks * 32 + hlf * 16 + e16;        // 0..1023
    unsigned gate = n >> 9;
    unsigned hc   = n & 511;
    const float* Wx[4] = {Wxi, Wxf, Wxo, Wxc};
    const float* Wh[4] = {Whi, Whf, Who, Whc};
    float v = (k < 512) ? Wx[gate][(size_t)k * Hsz + hc]
                        : Wh[gate][(size_t)(k - 512) * Hsz + hc];
    Wpack[tid] = (__bf16)v;
}

// ---------------------------- zero initial state ----------------------------
__global__ __launch_bounds__(256) void init_state(float* __restrict__ cst,
                                                  __bf16* __restrict__ h0) {
    int i = blockIdx.x * 256 + threadIdx.x;          // 32768
    cst[i] = 0.0f;
    h0[i]  = (__bf16)0.0f;
}

// ------------------------------ one LSTM step -------------------------------
// grid = 32 (h-column tiles of 16), block = 512 (16 waves).
// wave w: gate = w>>2, mTile = w&3 -> one 16x16 gate tile, K=1024 in 32 WMMAs.
__global__ __launch_bounds__(512) void lstm_step(
    const __bf16* __restrict__ Xbf,   const __bf16* __restrict__ Wpack,
    const __bf16* __restrict__ hprev, __bf16* __restrict__ hnext,
    float* __restrict__ cst,
    const float* __restrict__ bi, const float* __restrict__ bfv,
    const float* __restrict__ bo, const float* __restrict__ bc,
    float* __restrict__ out, int t) {
    __shared__ float lds[4][Bsz][16];                // 16 KB

    const int tid   = threadIdx.x;
    const int lane  = tid & 31;
    const int wave  = tid >> 5;
    const int gate  = wave >> 2;
    const int mTile = wave & 3;
    const int ct    = blockIdx.x;                    // 0..31
    const int hlf   = lane >> 4;
    const int mrow  = mTile * 16 + (lane & 15);      // batch row 0..63

    const int ntg = gate * 32 + ct;                  // global n-tile 0..127
    const __bf16* wbase = Wpack + (size_t)ntg * (32 * 32 * 16);
    const __bf16* xrow  = Xbf + ((size_t)mrow * Ssz + t) * Dsz;
    const __bf16* hrow  = hprev + (size_t)mrow * Hsz;

    v8f acc = {};

    // K = 0..511 : x_t part
#pragma unroll 4
    for (int ks = 0; ks < 16; ++ks) {
        union { uint4 u[2]; v16bf v; } a, b;
        const __bf16* p = xrow + ks * 32 + hlf * 16;
        a.u[0] = *(const uint4*)(p);
        a.u[1] = *(const uint4*)(p + 8);
        const __bf16* q = wbase + ((size_t)ks * 32 + lane) * 16;
        b.u[0] = *(const uint4*)(q);
        b.u[1] = *(const uint4*)(q + 8);
        acc = __builtin_amdgcn_wmma_f32_16x16x32_bf16(
            false, a.v, false, b.v, (short)0, acc, false, false);
    }
    // K = 512..1023 : h_{t-1} part
#pragma unroll 4
    for (int ks = 16; ks < 32; ++ks) {
        union { uint4 u[2]; v16bf v; } a, b;
        const __bf16* p = hrow + (ks - 16) * 32 + hlf * 16;
        a.u[0] = *(const uint4*)(p);
        a.u[1] = *(const uint4*)(p + 8);
        const __bf16* q = wbase + ((size_t)ks * 32 + lane) * 16;
        b.u[0] = *(const uint4*)(q);
        b.u[1] = *(const uint4*)(q + 8);
        acc = __builtin_amdgcn_wmma_f32_16x16x32_bf16(
            false, a.v, false, b.v, (short)0, acc, false, false);
    }

    // C/D layout: VGPR r -> M = r + half*8, N = lane&15
    {
        const int m0 = mTile * 16 + hlf * 8;
        const int n  = lane & 15;
#pragma unroll
        for (int r = 0; r < 8; ++r)
            lds[gate][m0 + r][n] = acc[r];
    }
    __syncthreads();

    // elementwise gate math: 64x16 = 1024 elements over 512 threads
#pragma unroll
    for (int i = tid; i < Bsz * 16; i += 512) {
        const int m  = i >> 4;
        const int n  = i & 15;
        const int nc = ct * 16 + n;
        float gi = lds[0][m][n] + bi[nc];
        float gf = lds[1][m][n] + bfv[nc];
        float go = lds[2][m][n] + bo[nc];
        float gc = lds[3][m][n] + bc[nc];
        float iv = 1.0f / (1.0f + __expf(-gi));
        float fv = 1.0f / (1.0f + __expf(-gf));
        float ov = 1.0f / (1.0f + __expf(-go));
        float gv = tanhf(gc);
        const int co = m * Hsz + nc;
        float cn = fv * cst[co] + iv * gv;
        float hv = ov * tanhf(cn);
        cst[co]   = cn;
        hnext[co] = (__bf16)hv;
        out[((size_t)m * Ssz + t) * Hsz + nc] = hv;
    }
}

// ------------------------- final h/c extraction -----------------------------
__global__ __launch_bounds__(256) void lstm_final(const float* __restrict__ out,
                                                  const float* __restrict__ cst,
                                                  float* __restrict__ hfin,
                                                  float* __restrict__ cfin) {
    int i = blockIdx.x * 256 + threadIdx.x;          // 32768
    int b = i >> 9, n = i & 511;
    hfin[i] = out[((size_t)b * Ssz + (Ssz - 1)) * Hsz + n];
    cfin[i] = cst[i];
}

// ---------------------------------------------------------------------------
extern "C" void kernel_launch(void* const* d_in, const int* in_sizes, int n_in,
                              void* d_out, int out_size, void* d_ws, size_t ws_size,
                              hipStream_t stream) {
    const float* X   = (const float*)d_in[0];
    const float* Wxi = (const float*)d_in[1];
    const float* Whi = (const float*)d_in[2];
    const float* bi  = (const float*)d_in[3];
    const float* Wxf = (const float*)d_in[4];
    const float* Whf = (const float*)d_in[5];
    const float* bfv = (const float*)d_in[6];
    const float* Wxo = (const float*)d_in[7];
    const float* Who = (const float*)d_in[8];
    const float* bo  = (const float*)d_in[9];
    const float* Wxc = (const float*)d_in[10];
    const float* Whc = (const float*)d_in[11];
    const float* bc  = (const float*)d_in[12];

    float* out  = (float*)d_out;
    float* hfin = out + (size_t)Bsz * Ssz * Hsz;
    float* cfin = hfin + (size_t)Bsz * Hsz;

    char* ws = (char*)d_ws;
    __bf16* Xbf   = (__bf16*)(ws);                    //  67,108,864 B
    __bf16* Wpack = (__bf16*)(ws + 67108864);         //   4,194,304 B
    float*  cst   = (float*) (ws + 71303168);         //     131,072 B
    __bf16* hb0   = (__bf16*)(ws + 71434240);         //      65,536 B
    __bf16* hb1   = (__bf16*)(ws + 71499776);         //      65,536 B

    // X -> bf16 (33.5M elems, 4 per thread)
    conv_bf16<<<32768, 256, 0, stream>>>(X, Xbf);
    // pack [Wx;Wh] into WMMA B-fragment order (2M elems)
    pack_w<<<8192, 256, 0, stream>>>(Wxi, Wxf, Wxo, Wxc, Whi, Whf, Who, Whc, Wpack);
    // h0 = c0 = 0
    init_state<<<128, 256, 0, stream>>>(cst, hb0);

    for (int t = 0; t < Ssz; ++t) {
        const __bf16* hp = (t & 1) ? hb1 : hb0;
        __bf16*       hn = (t & 1) ? hb0 : hb1;
        lstm_step<<<32, 512, 0, stream>>>(Xbf, Wpack, hp, hn, cst,
                                          bi, bfv, bo, bc, out, t);
    }

    lstm_final<<<128, 256, 0, stream>>>(out, cst, hfin, cfin);
}
